// Attention_17248588661289
// MI455X (gfx1250) — compile-verified, matrix operand order checked by reading, not run on hardware
//
#include <hip/hip_runtime.h>

typedef _Float16 f16;
typedef __attribute__((ext_vector_type(4)))  _Float16 v4h;
typedef __attribute__((ext_vector_type(8)))  _Float16 v8h;
typedef __attribute__((ext_vector_type(16))) _Float16 v16h;
typedef __attribute__((ext_vector_type(8)))  float    v8f;

#define BB   2
#define SS   2048
#define DD   2048
#define HH   16
#define KVH  4
#define HDIM 128
#define GG   (HH / KVH)

// ---------------------------------------------------------------------------
// Tensor Data Mover support (guarded; falls back to vector staging if absent)
// ---------------------------------------------------------------------------
#if defined(__has_builtin)
#if __has_builtin(__builtin_amdgcn_tensor_load_to_lds)
#define HAVE_TDM 1
#endif
#endif

#ifdef HAVE_TDM
typedef __attribute__((ext_vector_type(4))) unsigned int u32x4;
typedef __attribute__((ext_vector_type(4))) int          i32x4;
typedef __attribute__((ext_vector_type(8))) int          i32x8;

// Issue an async TDM load of a 128(row) x 32(k) f16 tile from a row-major
// [Mt x Kt] tensor (row stride Kt elements) into LDS at lds_byte_off.
// LDS padding: every 16 dwords (one 64B tile row) insert 4 dwords (16B)
// -> LDS row stride 80B == f16[40], matching the As[128][40] layout.
// D# layout per CDNA5 ISA §8.3 (group 0) / §8.4 (group 1); groups 2,3 unused.
__device__ __forceinline__
void tdm_load_A_tile(const f16* gsrc, unsigned lds_byte_off, int Kt, int Mt)
{
    const unsigned long long ga = (unsigned long long)(size_t)gsrc;
    u32x4 g0;
    g0[0] = 1u;                                              // count=1 (user mode)
    g0[1] = lds_byte_off;                                    // lds_addr
    g0[2] = (unsigned)ga;                                    // global_addr[31:0]
    g0[3] = (unsigned)((ga >> 32) & 0x01FFFFFFu)             // global_addr[56:32]
          | 0x80000000u;                                     // type=2 ("image")
    i32x8 g1;
    g1[0] = (1 << 16)                                        // data_size = 2B
          | (1 << 20)                                        // pad_enable
          | (3 << 22)                                        // pad_interval = 16 dwords
          | (3 << 25);                                       // pad_amount  = 4 dwords
    g1[1] = (Kt & 0xFFFF) << 16;                             // tensor_dim0[15:0]
    g1[2] = ((Kt >> 16) & 0xFFFF) | ((Mt & 0xFFFF) << 16);   // dim0[31:16] | dim1[15:0]
    g1[3] = ((Mt >> 16) & 0xFFFF) | (32 << 16);              // dim1[31:16] | tile_dim0=32
    g1[4] = 128;                                             // tile_dim1=128, tile_dim2=0
    g1[5] = Kt;                                              // tensor_dim0_stride[31:0]
    g1[6] = 0;                                               // stride0[47:32] | stride1 lo
    g1[7] = 0;
    i32x4 z4 = {0, 0, 0, 0};
#if __clang_major__ >= 23
    i32x8 z8 = {0, 0, 0, 0, 0, 0, 0, 0};
    __builtin_amdgcn_tensor_load_to_lds(g0, g1, z4, z4, z8, 0);
#else
    __builtin_amdgcn_tensor_load_to_lds(g0, g1, z4, z4, 0);
#endif
}
#endif // HAVE_TDM

__device__ __forceinline__ void wait_tensorcnt_1()
{
#if defined(__has_builtin)
#if __has_builtin(__builtin_amdgcn_s_wait_tensorcnt)
    __builtin_amdgcn_s_wait_tensorcnt(1);
#else
    asm volatile("s_wait_tensorcnt 0x1" ::: "memory");
#endif
#endif
}
__device__ __forceinline__ void wait_tensorcnt_0()
{
#if defined(__has_builtin)
#if __has_builtin(__builtin_amdgcn_s_wait_tensorcnt)
    __builtin_amdgcn_s_wait_tensorcnt(0);
#else
    asm volatile("s_wait_tensorcnt 0x0" ::: "memory");
#endif
#endif
}

// ---------------------------------------------------------------------------
// One-time f32 -> f16 conversion (x and all weights).
// ---------------------------------------------------------------------------
__global__ __launch_bounds__(256)
void cvt_f16(const float* __restrict__ src, f16* __restrict__ dst, const int n4)
{
    const int i = blockIdx.x * 256 + threadIdx.x;
    if (i < n4) {
        const float4 v = ((const float4*)src)[i];
        v4h o;
        o[0] = (f16)v.x; o[1] = (f16)v.y; o[2] = (f16)v.z; o[3] = (f16)v.w;
        ((v4h*)dst)[i] = o;
    }
}

// ---------------------------------------------------------------------------
// WMMA GEMM: C[M,N] = A[M,K] * Bw[K,N], both f16 row-major.
// 256 threads = 8 waves, tile 128x128x32; wave = 32x64 quadrant (8 wmma/K-step).
// A tile streamed into double-buffered LDS by the Tensor Data Mover (async,
// TENSORcnt-tracked), overlapping the WMMA work on the previous tile.
// B stored transposed in LDS so both operands read as two ds_load_b128.
// Optional fused interleaved-RoPE epilogue.
// ---------------------------------------------------------------------------
template <typename OT, bool ROPE>
__global__ __launch_bounds__(256)
void gemm_wmma(const f16* __restrict__ A, const f16* __restrict__ Bw,
               OT* __restrict__ C, const int M, const int N, const int K,
               const float* __restrict__ fcos, const float* __restrict__ fsin)
{
    __shared__ __align__(16) f16 As[2][128][40];   // rows x k (32 used, stride 40)
    __shared__ __align__(16) f16 Bs[128][40];      // cols x k (transposed)

    const int tid  = threadIdx.x;
    const int lane = tid & 31;
    const int wave = tid >> 5;       // 0..7
    const int wm   = wave >> 1;      // 0..3 : 32-row slab
    const int wn   = wave & 1;       // 0..1 : 64-col slab
    const int half = lane >> 4;
    const int l16  = lane & 15;
    const int m0   = blockIdx.y * 128;
    const int n0   = blockIdx.x * 128;

    v8f acc[2][4] = {};

#ifdef HAVE_TDM
    int buf = 0;
    if (wave == 0)   // one TDM issue per workgroup (EXEC-independent DMA)
        tdm_load_A_tile(A + (size_t)m0 * K, (unsigned)(size_t)&As[0][0][0], K, M);
#else
    const int buf = 0;
#endif

    for (int kk = 0; kk < K; kk += 32) {
        // ---- stage B tile (32x128) transposed into Bs[col][k] -----------
        #pragma unroll
        for (int p = 0; p < 2; ++p) {
            const int k  = (tid >> 4) + p * 16;
            const int cg = (tid & 15) * 8;
            const v8h bv = *(const v8h*)(Bw + (size_t)(kk + k) * N + n0 + cg);
            #pragma unroll
            for (int j = 0; j < 8; ++j) Bs[cg + j][k] = bv[j];
        }
        if (kk + 32 < K)
            __builtin_prefetch(Bw + (size_t)(kk + 32 + (tid >> 4)) * N + n0 + (tid & 15) * 8, 0, 1);

#ifdef HAVE_TDM
        // ---- A tile: async TDM double-buffer ----------------------------
        if (wave == 0) {
            if (kk + 32 < K) {
                tdm_load_A_tile(A + (size_t)m0 * K + (kk + 32),
                                (unsigned)(size_t)&As[buf ^ 1][0][0], K, M);
                wait_tensorcnt_1();        // current tile complete, next in flight
            } else {
                wait_tensorcnt_0();
            }
        }
#else
        // ---- A tile: synchronous vector staging -------------------------
        #pragma unroll
        for (int p = 0; p < 2; ++p) {
            const int row = (tid >> 2) + p * 64;
            const int cg  = (tid & 3) * 8;
            *(v8h*)&As[0][row][cg] =
                *(const v8h*)(A + (size_t)(m0 + row) * K + kk + cg);
        }
#endif
        __syncthreads();

        // ---- operand fragments: two ds_load_b128 each -------------------
        v16h af[2], bf[4];
        #pragma unroll
        for (int mi = 0; mi < 2; ++mi) {
            const int row = wm * 32 + mi * 16 + l16;
            const v8h lo = *(const v8h*)&As[buf][row][half * 8];
            const v8h hi = *(const v8h*)&As[buf][row][16 + half * 8];
            af[mi] = __builtin_shufflevector(lo, hi, 0,1,2,3,4,5,6,7,
                                                     8,9,10,11,12,13,14,15);
        }
        #pragma unroll
        for (int ni = 0; ni < 4; ++ni) {
            const int col = wn * 64 + ni * 16 + l16;
            const v8h lo = *(const v8h*)&Bs[col][half * 8];
            const v8h hi = *(const v8h*)&Bs[col][16 + half * 8];
            bf[ni] = __builtin_shufflevector(lo, hi, 0,1,2,3,4,5,6,7,
                                                     8,9,10,11,12,13,14,15);
        }
        #pragma unroll
        for (int mi = 0; mi < 2; ++mi)
            #pragma unroll
            for (int ni = 0; ni < 4; ++ni)
                acc[mi][ni] = __builtin_amdgcn_wmma_f32_16x16x32_f16(
                    false, af[mi], false, bf[ni], (short)0, acc[mi][ni],
                    false, false);
        __syncthreads();
#ifdef HAVE_TDM
        buf ^= 1;
#endif
    }

    // ---- epilogue: optional fused RoPE, then store ---------------------
    #pragma unroll
    for (int mi = 0; mi < 2; ++mi)
        #pragma unroll
        for (int ni = 0; ni < 4; ++ni)
            #pragma unroll
            for (int r = 0; r < 8; ++r) {
                const int row = m0 + wm * 32 + mi * 16 + half * 8 + r;
                const int col = n0 + wn * 64 + ni * 16 + l16;
                float v = acc[mi][ni][r];
                if constexpr (ROPE) {
                    const float pv = __shfl_xor(v, 1, 32);  // partner column
                    const int s  = row % SS;                // row = b*S + s
                    const int hd = col & (HDIM - 1);
                    const int j  = hd >> 1;
                    const float cv = fcos[s * (HDIM / 2) + j];
                    const float sv = fsin[s * (HDIM / 2) + j];
                    v = (hd & 1) ? (pv * sv + v * cv)       // odd: xe*s + xo*c
                                 : (v * cv - pv * sv);      // even: xe*c - xo*s
                }
                C[(size_t)row * N + col] = (OT)v;
            }
}

// ---------------------------------------------------------------------------
// Flash-attention (causal, GQA). One wave = one 16-row Q tile; 32-key chunks;
// Q@K^T and P@V on WMMA, online softmax in f32.
// ---------------------------------------------------------------------------
__global__ __launch_bounds__(256)
void attn_wmma(const f16* __restrict__ q_ws, const f16* __restrict__ k_ws,
               const f16* __restrict__ v_ws, f16* __restrict__ a_ws)
{
    __shared__ __align__(16) f16 P[8][16][36];

    const int tid  = threadIdx.x;
    const int lane = tid & 31;
    const int wave = tid >> 5;
    const int half = lane >> 4;
    const int l16  = lane & 15;

    const int bh = blockIdx.y;
    const int b  = bh / HH;
    const int h  = bh % HH;
    const int kv = h / GG;
    const int qs0 = blockIdx.x * 128 + wave * 16;
    const float scale = 0.08838834764831845f;    // 1/sqrt(128)

    v16h qf[4];
    const f16* qbase = q_ws + ((size_t)(b * SS + qs0 + l16)) * (HH * HDIM) + h * HDIM;
    #pragma unroll
    for (int t = 0; t < 4; ++t)
        #pragma unroll
        for (int e = 0; e < 16; ++e) {
            const int d = t * 32 + ((e >> 3) << 4) + (half << 3) + (e & 7);
            qf[t][e] = qbase[d];
        }

    float m[8], l[8];
    v8f o[8] = {};
    #pragma unroll
    for (int r = 0; r < 8; ++r) { m[r] = -1.0e30f; l[r] = 0.0f; }

    const int nch = (blockIdx.x * 128 + 128) / 32;   // block-uniform
    for (int ch = 0; ch < nch; ++ch) {
        const int kb = ch * 32;

        v8f s0 = {}, s1 = {};
        #pragma unroll
        for (int t = 0; t < 4; ++t) {
            v16h kf0, kf1;
            #pragma unroll
            for (int e = 0; e < 16; ++e) {
                const int d = t * 32 + ((e >> 3) << 4) + (half << 3) + (e & 7);
                kf0[e] = k_ws[((size_t)(b * SS + kb + l16))      * (KVH * HDIM) + kv * HDIM + d];
                kf1[e] = k_ws[((size_t)(b * SS + kb + 16 + l16)) * (KVH * HDIM) + kv * HDIM + d];
            }
            s0 = __builtin_amdgcn_wmma_f32_16x16x32_f16(false, qf[t], false, kf0,
                                                        (short)0, s0, false, false);
            s1 = __builtin_amdgcn_wmma_f32_16x16x32_f16(false, qf[t], false, kf1,
                                                        (short)0, s1, false, false);
        }

        #pragma unroll
        for (int r = 0; r < 8; ++r) {
            const int row = qs0 + half * 8 + r;
            float a0 = s0[r] * scale + ((kb + l16)      > row ? -1.0e9f : 0.0f);
            float a1 = s1[r] * scale + ((kb + 16 + l16) > row ? -1.0e9f : 0.0f);
            float mx = fmaxf(a0, a1);
            #pragma unroll
            for (int msk = 1; msk < 16; msk <<= 1)
                mx = fmaxf(mx, __shfl_xor(mx, msk, 32));
            const float mn = fmaxf(m[r], mx);
            const float sc = __expf(m[r] - mn);
            const float p0 = __expf(a0 - mn);
            const float p1 = __expf(a1 - mn);
            float rs = p0 + p1;
            #pragma unroll
            for (int msk = 1; msk < 16; msk <<= 1)
                rs += __shfl_xor(rs, msk, 32);
            l[r] = l[r] * sc + rs;
            m[r] = mn;
            #pragma unroll
            for (int nt = 0; nt < 8; ++nt) o[nt][r] *= sc;
            P[wave][half * 8 + r][l16]      = (f16)p0;
            P[wave][half * 8 + r][16 + l16] = (f16)p1;
        }
        __syncthreads();

        v16h pf;
        #pragma unroll
        for (int e = 0; e < 16; ++e) {
            const int k = ((e >> 3) << 4) + (half << 3) + (e & 7);
            pf[e] = P[wave][l16][k];
        }

        #pragma unroll
        for (int nt = 0; nt < 8; ++nt) {
            v16h vf;
            #pragma unroll
            for (int e = 0; e < 16; ++e) {
                const int sk = ((e >> 3) << 4) + (half << 3) + (e & 7);
                vf[e] = v_ws[((size_t)(b * SS + kb + sk)) * (KVH * HDIM)
                             + kv * HDIM + nt * 16 + l16];
            }
            o[nt] = __builtin_amdgcn_wmma_f32_16x16x32_f16(false, pf, false, vf,
                                                           (short)0, o[nt],
                                                           false, false);
        }
        __syncthreads();
    }

    #pragma unroll
    for (int nt = 0; nt < 8; ++nt)
        #pragma unroll
        for (int r = 0; r < 8; ++r) {
            const int row = qs0 + half * 8 + r;
            a_ws[((size_t)(b * SS + row)) * (HH * HDIM) + h * HDIM + nt * 16 + l16]
                = (f16)(o[nt][r] / l[r]);
        }
}

// ---------------------------------------------------------------------------
extern "C" void kernel_launch(void* const* d_in, const int* in_sizes, int n_in,
                              void* d_out, int out_size, void* d_ws, size_t ws_size,
                              hipStream_t stream)
{
    const float* x    = (const float*)d_in[0];
    const float* fcos = (const float*)d_in[1];
    const float* fsin = (const float*)d_in[2];
    // d_in[3] = mask: unused (causality computed inline, same -1e9 additive mask)
    const float* wq   = (const float*)d_in[4];
    const float* wk   = (const float*)d_in[5];
    const float* wv   = (const float*)d_in[6];
    const float* wo   = (const float*)d_in[7];
    float* out = (float*)d_out;

    const size_t nx  = (size_t)BB * SS * DD;
    const size_t nwq = (size_t)DD * HH * HDIM;
    const size_t nwk = (size_t)DD * KVH * HDIM;
    const size_t nwo = (size_t)HH * HDIM * DD;

    f16* xh  = (f16*)d_ws;
    f16* wqh = xh  + nx;
    f16* wkh = wqh + nwq;
    f16* wvh = wkh + nwk;
    f16* woh = wvh + nwk;
    f16* q_ws = woh + nwo;                            // (B,S,H*HD)
    f16* k_ws = q_ws + (size_t)BB * SS * HH  * HDIM;  // (B,S,KV*HD)
    f16* v_ws = k_ws + (size_t)BB * SS * KVH * HDIM;
    f16* a_ws = v_ws + (size_t)BB * SS * KVH * HDIM;  // (B,S,H*HD)

    // 1) one-time downconversion of activations + weights
    cvt_f16<<<dim3(nx  / 1024), dim3(256), 0, stream>>>(x,  xh,  (int)(nx  / 4));
    cvt_f16<<<dim3(nwq / 1024), dim3(256), 0, stream>>>(wq, wqh, (int)(nwq / 4));
    cvt_f16<<<dim3(nwk / 1024), dim3(256), 0, stream>>>(wk, wkh, (int)(nwk / 4));
    cvt_f16<<<dim3(nwk / 1024), dim3(256), 0, stream>>>(wv, wvh, (int)(nwk / 4));
    cvt_f16<<<dim3(nwo / 1024), dim3(256), 0, stream>>>(wo, woh, (int)(nwo / 4));

    const int M = BB * SS;
    dim3 blk(256);

    // 2) Q = rope(x@wq), K = rope(x@wk), V = x@wv
    gemm_wmma<f16, true ><<<dim3((HH  * HDIM) / 128, M / 128), blk, 0, stream>>>(
        xh, wqh, q_ws, M, HH * HDIM, DD, fcos, fsin);
    gemm_wmma<f16, true ><<<dim3((KVH * HDIM) / 128, M / 128), blk, 0, stream>>>(
        xh, wkh, k_ws, M, KVH * HDIM, DD, fcos, fsin);
    gemm_wmma<f16, false><<<dim3((KVH * HDIM) / 128, M / 128), blk, 0, stream>>>(
        xh, wvh, v_ws, M, KVH * HDIM, DD, nullptr, nullptr);

    // 3) causal GQA flash attention
    attn_wmma<<<dim3(SS / 128, BB * HH), dim3(256), 0, stream>>>(q_ws, k_ws, v_ws, a_ws);

    // 4) out = attn @ wo  (f32 output)
    gemm_wmma<float, false><<<dim3(DD / 128, M / 128), blk, 0, stream>>>(
        a_ws, woh, out, M, DD, HH * HDIM, nullptr, nullptr);
}
